// MergeMultimodalLSTM_57698590654831
// MI455X (gfx1250) — matrix operand order repaired and checked by reading, compiled, not run on hardware
//
#include <hip/hip_runtime.h>
#include <hip/hip_bf16.h>
#include <math.h>

typedef __attribute__((ext_vector_type(16))) __bf16 v16bf;
typedef __attribute__((ext_vector_type(8)))  __bf16 v8bf;
typedef __attribute__((ext_vector_type(8)))  float  v8f;

#define NF      128
#define SPATIAL 1024        // 32*32
#define NOUT    512         // 4*NF
#define BATCH   4

// LDS slab geometry: [row 0..2][xcol 0..33][chan 0..31], chan innermost,
// row stride padded to 40 elems (80 B) => every 8-chan run is 16B-aligned
// and staging writes spread across banks.
#define CSTR 40

// ---------------------------------------------------------------------------
// Weight pack: W (512, Cin, 3, 3) f32  ->  Wp bf16 in B-fragment-native layout.
// tile = (pos*nChunk + cc)*32 + ntile   (pos = ky*3+kx, cc = K-chunk, ntile = oc/16)
// within tile: lane L (0..31), slot j (0..15):
//   oc = ntile*16 + (L&15);  k = (L>>4)*16 + j;  c = cc*32 + k
// Each lane later loads its 16 bf16 (32B) contiguously (2x global_load_b128).
// ---------------------------------------------------------------------------
__global__ void pack_weights(const float* __restrict__ W, unsigned short* __restrict__ Wp_,
                             int Cin, int total)
{
    __bf16* Wp = (__bf16*)Wp_;
    int idx = blockIdx.x * blockDim.x + threadIdx.x;
    if (idx >= total) return;
    int nChunk = Cin >> 5;
    int j     = idx & 15;
    int L     = (idx >> 4) & 31;
    int tile  = idx >> 9;
    int ntile = tile & 31;
    int cc    = (tile >> 5) % nChunk;
    int pos   = tile / (32 * nChunk);
    int ky = pos / 3, kx = pos - ky * 3;
    int oc = ntile * 16 + (L & 15);
    int c  = cc * 32 + ((L >> 4) * 16 + j);
    float w = W[(((size_t)oc * Cin + c) * 3 + ky) * 3 + kx];
    Wp[idx] = (__bf16)w;
}

__global__ void zero_f32(float* __restrict__ p, int n)
{
    int i = blockIdx.x * blockDim.x + threadIdx.x;
    if (i < n) p[i] = 0.0f;
}

// ---------------------------------------------------------------------------
// Implicit-GEMM conv: 3x3 conv as 9 shifted GEMMs over channels (bf16 WMMA,
// f32 accumulate). Block: 256 threads = 8 waves. Block tile: M=32 (one full
// image row, fixed b,y) x N=512 (all output channels). Wave w covers
// N = w*64 .. w*64+63 with a 2(M) x 4(N) grid of 16x16 WMMA accumulators.
// A-fragments: 2x ds_load_b128 per fragment from a channel-innermost LDS slab.
// B-fragments: 2x global_load_b128 per fragment from pre-packed bf16 weights.
// ---------------------------------------------------------------------------
__global__ __launch_bounds__(256)
void conv_gemm_bf16(const float* __restrict__ s0, const float* __restrict__ s1,
                    const float* __restrict__ s2,
                    int bs0, int bs1, int bs2,
                    const unsigned short* __restrict__ Wp_,
                    float* __restrict__ z, int Cin)
{
    const __bf16* Wp = (const __bf16*)Wp_;
    __shared__ alignas(16) __bf16 Abuf[3 * 34 * CSTR];   // 8160 B

    const int tid  = threadIdx.x;
    const int lane = tid & 31;
    const int wv   = tid >> 5;                 // wave 0..7
    const int mtile = blockIdx.x;              // 0..127 -> (b, y)
    const int b = mtile >> 5;
    const int y = mtile & 31;
    const int n0 = wv * 64;                    // output-channel base for wave
    const int nChunk = Cin >> 5;

    v8f acc[2][4];
#pragma unroll
    for (int i = 0; i < 2; ++i)
#pragma unroll
        for (int j = 0; j < 4; ++j)
#pragma unroll
            for (int k = 0; k < 8; ++k) acc[i][j][k] = 0.0f;

    // A-fragment lane mapping used here (contiguous K, matches the Wp pack):
    //   lanes 0-15 : row M = lane,    K = 0..15
    //   lanes 16-31: row M = lane-16, K = 16..31
    const int mrow = lane & 15;
    const int kb   = (lane >> 4) * 16;

    for (int cc = 0; cc < nChunk; ++cc) {
        __syncthreads();   // protect previous iteration's LDS reads
        // ---- stage 3 rows x 34 cols (haloed) x 32 chans into LDS as bf16 ----
        // Global read keeps x innermost (coalesced); LDS write is chan-innermost.
        for (int e = tid; e < 3 * 32 * 34; e += 256) {
            int xcol = e % 34;
            int t2   = e / 34;
            int c    = t2 & 31;
            int r    = t2 >> 5;
            int xx = xcol - 1;
            int yy = y + r - 1;
            float v = 0.0f;
            if ((unsigned)xx < 32u && (unsigned)yy < 32u) {
                int cg = cc * 32 + c;
                const float* sp; int bs;
                if (cg < 128)      { sp = s0; bs = bs0; }
                else if (cg < 256) { sp = s1; bs = bs1; cg -= 128; }
                else               { sp = s2; bs = bs2; cg -= 256; }
                v = sp[(size_t)b * bs + (size_t)cg * SPATIAL + yy * 32 + xx];
            }
            Abuf[(r * 34 + xcol) * CSTR + c] = (__bf16)v;
        }
        __syncthreads();

        // ---- 9 kernel taps: shifted GEMM over this 32-channel chunk ----
#pragma unroll
        for (int pos = 0; pos < 9; ++pos) {
            const int ky = pos / 3;
            const int kx = pos - ky * 3;

            v16bf a[2];
#pragma unroll
            for (int mt = 0; mt < 2; ++mt) {
                const int xb = mt * 16 + mrow + kx;           // shifted col 0..33
                const __bf16* ap = &Abuf[(ky * 34 + xb) * CSTR + kb];
                v8bf lo = *(const v8bf*)ap;                   // K kb..kb+7
                v8bf hi = *(const v8bf*)(ap + 8);             // K kb+8..kb+15
                a[mt] = __builtin_shufflevector(lo, hi,
                        0, 1, 2, 3, 4, 5, 6, 7, 8, 9, 10, 11, 12, 13, 14, 15);
            }

            const __bf16* bp = Wp
                + (((size_t)(pos * nChunk + cc) * 32 + (n0 >> 4)) * 512)
                + lane * 16;
            v16bf bf[4];
#pragma unroll
            for (int nt = 0; nt < 4; ++nt)
                bf[nt] = *(const v16bf*)(bp + nt * 512);

#pragma unroll
            for (int mt = 0; mt < 2; ++mt)
#pragma unroll
                for (int nt = 0; nt < 4; ++nt)
                    acc[mt][nt] = __builtin_amdgcn_wmma_f32_16x16x32_bf16(
                        false, a[mt], false, bf[nt], (short)0, acc[mt][nt],
                        false, false);
        }
    }

    // ---- epilogue: C/D layout -> z (B, 512, 32, 32) f32, contiguous stores ----
    // VGPR r of lanes 0-15: (M=r, N=lane); lanes 16-31: (M=8+r, N=lane-16).
    // M maps to x, so the 8 accumulator floats are 8 consecutive x.
    const int lh = lane >> 4;
    const int nl = lane & 15;
#pragma unroll
    for (int mt = 0; mt < 2; ++mt) {
#pragma unroll
        for (int nt = 0; nt < 4; ++nt) {
            int oc = n0 + nt * 16 + nl;
            float* dst = z + (((size_t)b * NOUT + oc) * 32 + y) * 32
                           + mt * 16 + lh * 8;
            v8f v = acc[mt][nt];
            float4 flo = make_float4(v[0], v[1], v[2], v[3]);
            float4 fhi = make_float4(v[4], v[5], v[6], v[7]);
            ((float4*)dst)[0] = flo;
            ((float4*)dst)[1] = fhi;
        }
    }
}

// ---------------------------------------------------------------------------
// Fused LSTM pointwise update:  z -> (c_next, h_next).  Gate order: i,f,o,g.
// ---------------------------------------------------------------------------
__device__ __forceinline__ float sigmoidf(float x) { return 1.0f / (1.0f + __expf(-x)); }

__global__ void lstm_pointwise(const float* __restrict__ z, const float* __restrict__ bias,
                               const float* __restrict__ cprev,
                               float* __restrict__ cnext, float* __restrict__ hnext)
{
    int idx = blockIdx.x * blockDim.x + threadIdx.x;   // over B*128*1024
    if (idx >= BATCH * NF * SPATIAL) return;
    int p  = idx & (SPATIAL - 1);
    int ch = (idx >> 10) & 127;
    int b  = idx >> 17;
    size_t base = ((size_t)b * NOUT + ch) * SPATIAL + p;
    float gi = z[base]                          + bias[ch];
    float gf = z[base + (size_t)128 * SPATIAL]  + bias[128 + ch];
    float go = z[base + (size_t)256 * SPATIAL]  + bias[256 + ch];
    float gg = z[base + (size_t)384 * SPATIAL]  + bias[384 + ch];
    float c = sigmoidf(gf) * cprev[idx] + sigmoidf(gi) * tanhf(gg);
    float h = sigmoidf(go) * tanhf(c);
    cnext[idx] = c;
    hnext[idx] = h;
}

// ---------------------------------------------------------------------------
extern "C" void kernel_launch(void* const* d_in, const int* in_sizes, int n_in,
                              void* d_out, int out_size, void* d_ws, size_t ws_size,
                              hipStream_t stream)
{
    const float* x1 = (const float*)d_in[0];  // (4,8,128,32,32)
    const float* x2 = (const float*)d_in[1];
    const float* W1 = (const float*)d_in[2];  // (512,384,3,3)
    const float* b1 = (const float*)d_in[3];
    const float* W2 = (const float*)d_in[4];  // (512,256,3,3)
    const float* b2 = (const float*)d_in[5];
    const float* W3 = (const float*)d_in[6];  // (512,256,3,3)
    const float* b3 = (const float*)d_in[7];
    float* h3 = (float*)d_out;                // (4,128,32,32) running h3 buffer

    // Workspace carve-up (256B aligned slices).
    char* ws = (char*)d_ws;
    size_t off = 0;
    auto take = [&](size_t bytes) -> void* {
        void* p = ws + off;
        off += (bytes + 255) & ~(size_t)255;
        return p;
    };
    const int totW1 = 9 * 384 * 512;
    const int totW2 = 9 * 256 * 512;
    unsigned short* Wp1 = (unsigned short*)take((size_t)totW1 * 2);
    unsigned short* Wp2 = (unsigned short*)take((size_t)totW2 * 2);
    unsigned short* Wp3 = (unsigned short*)take((size_t)totW2 * 2);
    float* zbuf  = (float*)take((size_t)BATCH * NOUT * SPATIAL * 4);
    const int NSTATE = BATCH * NF * SPATIAL;          // 524288
    float* state = (float*)take((size_t)5 * NSTATE * 4);
    float* h1 = state;
    float* c1 = state + 1 * NSTATE;
    float* h2 = state + 2 * NSTATE;
    float* c2 = state + 3 * NSTATE;
    float* c3 = state + 4 * NSTATE;

    // 1) pack weights to bf16 WMMA-B layout
    pack_weights<<<(totW1 + 255) / 256, 256, 0, stream>>>(W1, Wp1, 384, totW1);
    pack_weights<<<(totW2 + 255) / 256, 256, 0, stream>>>(W2, Wp2, 256, totW2);
    pack_weights<<<(totW2 + 255) / 256, 256, 0, stream>>>(W3, Wp3, 256, totW2);

    // 2) zero initial h/c state (h3 lives in d_out)
    zero_f32<<<(5 * NSTATE + 255) / 256, 256, 0, stream>>>(state, 5 * NSTATE);
    zero_f32<<<(NSTATE + 255) / 256, 256, 0, stream>>>(h3, NSTATE);

    // 3) recurrent sweep
    const int bsX = 8 * NF * SPATIAL;   // batch stride of x1/x2 (T folded in)
    const int bsH = NF * SPATIAL;       // batch stride of state buffers
    const int pwBlocks = (NSTATE + 255) / 256;
    for (int t = 0; t < 8; ++t) {
        const float* x1t = x1 + (size_t)t * NF * SPATIAL;
        const float* x2t = x2 + (size_t)t * NF * SPATIAL;
        // layer 1: input = [x1_t, x2_t, h1]  (Cin = 384)
        conv_gemm_bf16<<<128, 256, 0, stream>>>(x1t, x2t, h1, bsX, bsX, bsH,
                                                Wp1, zbuf, 384);
        lstm_pointwise<<<pwBlocks, 256, 0, stream>>>(zbuf, b1, c1, c1, h1);
        // layer 2: input = [h1, h2]  (Cin = 256)
        conv_gemm_bf16<<<128, 256, 0, stream>>>(h1, h2, h1, bsH, bsH, bsH,
                                                Wp2, zbuf, 256);
        lstm_pointwise<<<pwBlocks, 256, 0, stream>>>(zbuf, b2, c2, c2, h2);
        // layer 3: input = [h2, h3]  (Cin = 256)
        conv_gemm_bf16<<<128, 256, 0, stream>>>(h2, h3, h2, bsH, bsH, bsH,
                                                Wp3, zbuf, 256);
        lstm_pointwise<<<pwBlocks, 256, 0, stream>>>(zbuf, b3, c3, c3, h3);
    }
}
